// Int8Linear_23905787969796
// MI455X (gfx1250) — 8-bit matrix operand order repaired and checked by reading, compiled, not run on hardware
//
#include <hip/hip_runtime.h>
#include <stdint.h>

#define BLK 128
#define QMAX 127.0f

typedef __attribute__((ext_vector_type(8))) int   v8i;
typedef __attribute__((ext_vector_type(4))) int   v4i;
typedef __attribute__((ext_vector_type(2))) int   v2i;
typedef __attribute__((ext_vector_type(4))) float v4f;

__device__ __forceinline__ float wave_amax32(float v) {
#pragma unroll
    for (int off = 16; off > 0; off >>= 1)
        v = fmaxf(v, __shfl_xor(v, off, 32));
    return v;
}

__device__ __forceinline__ int quant1(float v, float r) {
    int qi = (int)rintf(v * r);
    qi = qi > 127 ? 127 : qi;
    qi = qi < -128 ? -128 : qi;
    return qi;
}

__device__ __forceinline__ int pack4(v4f v, float r) {
    int q0 = quant1(v.x, r), q1 = quant1(v.y, r), q2 = quant1(v.z, r), q3 = quant1(v.w, r);
    return (q0 & 255) | ((q1 & 255) << 8) | ((q2 & 255) << 16) | ((q3 & 255) << 24);
}

__global__ __launch_bounds__(256) void act_quant_kernel(
    const float* __restrict__ x, int8_t* __restrict__ xq,
    float* __restrict__ xs_t, int M, int K) {
    const int Kb   = K / BLK;
    const int lane = threadIdx.x & 31;
    const int wave = threadIdx.x >> 5;
    const int blocksPerRow = Kb / 8;
    const int row = blockIdx.x / blocksPerRow;
    const int kb  = (blockIdx.x % blocksPerRow) * 8 + wave;

    const float* px = x + (size_t)row * K + kb * BLK + lane * 4;
    v4f xv = *(const v4f*)px;
    float a = fmaxf(fmaxf(fabsf(xv.x), fabsf(xv.y)), fmaxf(fabsf(xv.z), fabsf(xv.w)));
    a = wave_amax32(a);
    float s = fmaxf(a * (1.0f / QMAX), 1e-12f);
    float r = 1.0f / s;

    int p = pack4(xv, r);
    int* xqw = (int*)xq;
    xqw[((size_t)row * K + kb * BLK) / 4 + lane] = p;
    if (lane == 0) xs_t[(size_t)kb * M + row] = s;
}

__global__ __launch_bounds__(256) void weight_quant_kernel(
    const float* __restrict__ w, int8_t* __restrict__ wq,
    float* __restrict__ ws, int N, int K) {
    __shared__ float red[256];
    __shared__ float s_scale;
    const int Kb  = K / BLK;
    const int nb  = blockIdx.x / Kb;
    const int kb  = blockIdx.x % Kb;
    const int t   = threadIdx.x;
    const int row = t >> 1;
    const int half = t & 1;

    const float* pw = w + (size_t)(nb * BLK + row) * K + kb * BLK + half * 64;
    v4f v[16];
    float a = 0.0f;
#pragma unroll
    for (int i = 0; i < 16; ++i) {
        v[i] = ((const v4f*)pw)[i];
        a = fmaxf(a, fmaxf(fmaxf(fabsf(v[i].x), fabsf(v[i].y)),
                           fmaxf(fabsf(v[i].z), fabsf(v[i].w))));
    }
    red[t] = a;
    __syncthreads();
#pragma unroll
    for (int off = 128; off > 0; off >>= 1) {
        if (t < off) red[t] = fmaxf(red[t], red[t + off]);
        __syncthreads();
    }
    if (t == 0) {
        float s = fmaxf(red[0] * (1.0f / QMAX), 1e-12f);
        s_scale = s;
        ws[nb * Kb + kb] = s;
    }
    __syncthreads();
    float r = 1.0f / s_scale;

    v4i* dst = (v4i*)(wq + (size_t)(nb * BLK + row) * K + kb * BLK + half * 64);
#pragma unroll
    for (int i = 0; i < 4; ++i) {
        v4i o;
        o.x = pack4(v[4 * i + 0], r);
        o.y = pack4(v[4 * i + 1], r);
        o.z = pack4(v[4 * i + 2], r);
        o.w = pack4(v[4 * i + 3], r);
        dst[i] = o;
    }
}

__global__ __launch_bounds__(256) void int8_gemm_kernel(
    const int8_t* __restrict__ xq, const int8_t* __restrict__ wq,
    const float* __restrict__ xs_t, const float* __restrict__ ws,
    const float* __restrict__ bias, float* __restrict__ out,
    int M, int N, int K) {
    const int Kb    = K / BLK;
    const int lane  = threadIdx.x & 31;
    const int wave  = threadIdx.x >> 5;
    const int mw    = wave & 3;
    const int nw    = wave >> 2;
    const int mtile = blockIdx.x * 128 + mw * 32;
    const int ntile = blockIdx.y * 64 + nw * 32;
    const int nb    = (blockIdx.y * 64) / BLK;
    const int arow  = lane & 15;
    const int khalf = lane >> 4;

    const int8_t* aptr[2];
    aptr[0] = xq + (size_t)(mtile + arow) * K + khalf * 16;
    aptr[1] = xq + (size_t)(mtile + 16 + arow) * K + khalf * 16;
    const int8_t* bptr[2];
    bptr[0] = wq + (size_t)(ntile + arow) * K + khalf * 16;
    bptr[1] = wq + (size_t)(ntile + 16 + arow) * K + khalf * 16;

    const v8i zero8 = {0, 0, 0, 0, 0, 0, 0, 0};
    float facc[2][2][8];
#pragma unroll
    for (int ms = 0; ms < 2; ++ms)
#pragma unroll
        for (int ns = 0; ns < 2; ++ns)
#pragma unroll
            for (int v = 0; v < 8; ++v) facc[ms][ns][v] = 0.0f;

    for (int kb = 0; kb < Kb; ++kb) {
        const int k0 = kb * BLK;

        if (kb + 1 < Kb) {
            __builtin_prefetch(aptr[0] + k0 + BLK, 0, 1);
            __builtin_prefetch(bptr[0] + k0 + BLK, 0, 1);
        }

        v8i A[2][2];
#pragma unroll
        for (int ms = 0; ms < 2; ++ms) {
            const int8_t* base = aptr[ms] + k0;
#pragma unroll
            for (int ks = 0; ks < 2; ++ks)
#pragma unroll
                for (int j = 0; j < 4; ++j) {
                    v2i d = *(const v2i*)(base + ks * 64 + (j >> 1) * 32 + (j & 1) * 8);
                    A[ms][ks][2 * j]     = d.x;
                    A[ms][ks][2 * j + 1] = d.y;
                }
        }
        v8i Bf[2][2];
#pragma unroll
        for (int ns = 0; ns < 2; ++ns) {
            const int8_t* base = bptr[ns] + k0;
#pragma unroll
            for (int ks = 0; ks < 2; ++ks) {
                v4i lo = *(const v4i*)(base + ks * 64);
                v4i hi = *(const v4i*)(base + ks * 64 + 32);
                Bf[ns][ks][0] = lo.x; Bf[ns][ks][1] = lo.y;
                Bf[ns][ks][2] = lo.z; Bf[ns][ks][3] = lo.w;
                Bf[ns][ks][4] = hi.x; Bf[ns][ks][5] = hi.y;
                Bf[ns][ks][6] = hi.z; Bf[ns][ks][7] = hi.w;
            }
        }

        v8i iacc[2][2];
#pragma unroll
        for (int ms = 0; ms < 2; ++ms)
#pragma unroll
            for (int ns = 0; ns < 2; ++ns) iacc[ms][ns] = zero8;
#pragma unroll
        for (int ks = 0; ks < 2; ++ks)
#pragma unroll
            for (int ms = 0; ms < 2; ++ms)
#pragma unroll
                for (int ns = 0; ns < 2; ++ns)
                    iacc[ms][ns] = __builtin_amdgcn_wmma_i32_16x16x64_iu8(
                        true, A[ms][ks], true, Bf[ns][ks], iacc[ms][ns],
                        false, false);

        const float wsv = ws[nb * Kb + kb];
#pragma unroll
        for (int ms = 0; ms < 2; ++ms) {
            const float* sp = xs_t + (size_t)kb * M + mtile + ms * 16 + khalf * 8;
            v4f s0 = *(const v4f*)sp;
            v4f s1 = *(const v4f*)(sp + 4);
            float sv[8] = {s0.x, s0.y, s0.z, s0.w, s1.x, s1.y, s1.z, s1.w};
#pragma unroll
            for (int ns = 0; ns < 2; ++ns)
#pragma unroll
                for (int v = 0; v < 8; ++v)
                    facc[ms][ns][v] += (float)iacc[ms][ns][v] * (sv[v] * wsv);
        }
    }

#pragma unroll
    for (int ns = 0; ns < 2; ++ns) {
        const int n = ntile + ns * 16 + arow;
        const float b = bias[n];
#pragma unroll
        for (int ms = 0; ms < 2; ++ms) {
            const int mbase = mtile + ms * 16 + khalf * 8;
#pragma unroll
            for (int v = 0; v < 8; ++v)
                out[(size_t)(mbase + v) * N + n] = facc[ms][ns][v] + b;
        }
    }
}

extern "C" void kernel_launch(void* const* d_in, const int* in_sizes, int n_in,
                              void* d_out, int out_size, void* d_ws, size_t ws_size,
                              hipStream_t stream) {
    const float* x    = (const float*)d_in[0];
    const float* w    = (const float*)d_in[1];
    const float* bias = (const float*)d_in[2];
    float* out = (float*)d_out;

    const int N  = in_sizes[2];
    const int K  = in_sizes[1] / N;
    const int M  = in_sizes[0] / K;
    const int Kb = K / BLK;
    const int Nb = N / BLK;

    uint8_t* base = (uint8_t*)d_ws;
    size_t off = 0;
    auto carve = [&](size_t bytes) -> void* {
        off = (off + 255) & ~(size_t)255;
        void* p = base + off;
        off += bytes;
        return p;
    };
    int8_t* xq   = (int8_t*)carve((size_t)M * K);
    int8_t* wq   = (int8_t*)carve((size_t)N * K);
    float*  xs_t = (float*)carve((size_t)Kb * M * sizeof(float));
    float*  wsc  = (float*)carve((size_t)Nb * Kb * sizeof(float));
    (void)ws_size; (void)n_in; (void)out_size;

    act_quant_kernel<<<M * (Kb / 8), 256, 0, stream>>>(x, xq, xs_t, M, K);
    weight_quant_kernel<<<Nb * Kb, 256, 0, stream>>>(w, wq, wsc, N, K);

    dim3 grid(M / 128, N / 64);
    int8_gemm_kernel<<<grid, 256, 0, stream>>>(xq, wq, xs_t, wsc, bias, out, M, N, K);
}
